// DINONewVq_90056874262837
// MI455X (gfx1250) — matrix operand order repaired and checked by reading, not compiled, not run on hardware
//
#include <hip/hip_runtime.h>

// ---------------------------------------------------------------------------
// VQ-VAE vector quantization for MI455X (gfx1250, wave32, WMMA).
//   z:        (32, 384, 28, 28) f32
//   codebook: (2048, 384)       f32
//   out:      z_q (32,384,28,28) f32  ++  q_loss (1) f32
//
// Distance GEMM in bf16 via v_wmma_f32_16x16x32_bf16 with fused argmin.
// B operands staged through LDS with global_load_async_to_lds_b128
// (ASYNCcnt-tracked double buffer), dual accumulator chains for WMMA ILP.
// ---------------------------------------------------------------------------

typedef __attribute__((ext_vector_type(16))) __bf16 v16bf;
typedef __attribute__((ext_vector_type(8)))  __bf16 v8bf;
typedef __attribute__((ext_vector_type(8)))  float  v8f;

#define DIMC   384         // channels / GEMM K-dim
#define KCODE  2048        // codebook entries
#define HW     784         // 28*28
#define BATCH  32
#define NTOK   25088       // 32*784 tokens
#define OUT_N  9633792     // 32*384*784 output elements
#define BETA   0.25f

#define TILE_BYTES 12288   // one B tile: 16 codebook rows * 384 ch * 2B

// ---- workspace layout (bytes) ---------------------------------------------
#define WS_ZT    0                      // bf16 z, token-major  (N*D*2 = 19,267,584)
#define WS_CB    19267584               // bf16 codebook        (K*D*2 =  1,572,864)
#define WS_E2    20840448               // f32 ||e||^2          (K*4   =      8,192)
#define WS_IDX   20848640               // int argmin per token (N*4   =    100,352)
#define WS_PART  20948992               // f32 loss partials    (37632*4 =  150,528)

// ---------------------------------------------------------------------------
// 1) codebook -> bf16 copy + per-row sum of squares (f32)
// ---------------------------------------------------------------------------
__global__ __launch_bounds__(128) void prep_codebook(const float* __restrict__ cb,
                                                     __bf16* __restrict__ cb_bf,
                                                     float* __restrict__ e2) {
  __shared__ float red[128];
  const int k = blockIdx.x;
  const float* row = cb + (size_t)k * DIMC;
  float s = 0.f;
  for (int c = threadIdx.x; c < DIMC; c += 128) {
    float v = row[c];
    cb_bf[(size_t)k * DIMC + c] = (__bf16)v;
    s += v * v;
  }
  red[threadIdx.x] = s;
  __syncthreads();
  for (int off = 64; off > 0; off >>= 1) {
    if (threadIdx.x < off) red[threadIdx.x] += red[threadIdx.x + off];
    __syncthreads();
  }
  if (threadIdx.x == 0) e2[k] = red[0];
}

// ---------------------------------------------------------------------------
// 2) z (b,c,h,w) f32  ->  zt (n, c) bf16, n = b*784 + h*28 + w.
//    LDS 32x32 tile transpose so both global sides are coalesced.
// ---------------------------------------------------------------------------
__global__ __launch_bounds__(256) void prep_z(const float* __restrict__ z,
                                              __bf16* __restrict__ zt) {
  __shared__ float tile[32][33];
  const int hwt = blockIdx.x;   // 25 tiles over hw (784)
  const int ct  = blockIdx.y;   // 12 tiles over c  (384)
  const int b   = blockIdx.z;   // 32
  const int tx = threadIdx.x;   // 0..31
  const int ty = threadIdx.y;   // 0..7
  const int hw_r = hwt * 32 + tx;
#pragma unroll
  for (int r = 0; r < 4; ++r) {
    const int cl = ty + 8 * r;
    float v = 0.f;
    if (hw_r < HW) v = z[((size_t)(b * DIMC + ct * 32 + cl)) * HW + hw_r];
    tile[cl][tx] = v;
  }
  __syncthreads();
#pragma unroll
  for (int r = 0; r < 4; ++r) {
    const int hwl = ty + 8 * r;
    const int hw  = hwt * 32 + hwl;
    if (hw < HW) {
      const int n = b * HW + hw;
      zt[(size_t)n * DIMC + ct * 32 + tx] = (__bf16)tile[tx][hwl];
    }
  }
}

// ---------------------------------------------------------------------------
// Async-stage one contiguous 12KB B tile (16 codebook rows) into LDS.
// 32 lanes x 16B x 24 iterations, ASYNCcnt-tracked, no VGPR round trip.
// ---------------------------------------------------------------------------
__device__ __forceinline__ void stage_tile(const __bf16* __restrict__ gsrc,
                                           unsigned lds_off, int lane) {
  unsigned long long ga = (unsigned long long)(const void*)gsrc + (unsigned)lane * 16u;
  unsigned lo = lds_off + (unsigned)lane * 16u;
#pragma unroll
  for (int i = 0; i < 24; ++i) {           // 24 * 512B = 12KB
    asm volatile("global_load_async_to_lds_b128 %0, %1, off"
                 :: "v"(lo), "v"(ga) : "memory");
    ga += 512ull;
    lo += 512u;
  }
}

// ---------------------------------------------------------------------------
// 3) Fused distance GEMM + argmin.
//    Block = 16 tokens, 256 threads = 8 waves. Wave w scans column tiles
//    colbase = it*128 + w*16, it = 0..15  (covers all 2048 codes).
//    A tile (16x384 bf16) hoisted into 12 v16bf regs, reused for all tiles.
//    B tiles double-buffered in LDS via async copies (wave-private buffers,
//    in-order ASYNCcnt => s_wait_asynccnt 24 keeps one tile in flight).
//    Dual accumulator chains (even/odd K-chunks) for WMMA pipeline ILP.
//    score = ||e||^2 - 2 * (z . e)   (||z||^2 constant per row -> dropped)
// ---------------------------------------------------------------------------
__global__ __launch_bounds__(256) void vq_argmin(const __bf16* __restrict__ zt,
                                                 const __bf16* __restrict__ cb,
                                                 const float* __restrict__ e2,
                                                 int* __restrict__ min_idx) {
  __shared__ __attribute__((aligned(32))) unsigned char bstage[8][2][TILE_BYTES];
  __shared__ float lds_s[8 * 16];
  __shared__ int   lds_i[8 * 16];

  const int rowbase = blockIdx.x * 16;
  const int wave = threadIdx.x >> 5;
  const int lane = threadIdx.x & 31;
  const int lrow = lane & 15;     // A: M index / B,C,D: N index
  const int kg   = lane >> 4;     // K-half select per ISA 7.12.2

  // LDS byte offsets of this wave's two stage buffers (low 32 bits of the
  // generic pointer are the LDS address per the flat-aperture rule).
  const unsigned lds_buf0 = (unsigned)(unsigned long long)(const void*)&bstage[wave][0][0];
  const unsigned lds_buf1 = (unsigned)(unsigned long long)(const void*)&bstage[wave][1][0];

  // ---- load + hold the A tile (contiguous K order per 32-elem chunk, matches B):
  //   lane half 0: K = 0..15 + 32*chunk ; half 1: K = 16..31
  v16bf A[12];
  const __bf16* zrow = zt + (size_t)(rowbase + lrow) * DIMC;
#pragma unroll
  for (int c = 0; c < 12; ++c) {
    A[c] = *(const v16bf*)(zrow + c * 32 + 16 * kg);
  }

  float best[8];
  int   bidx[8];
#pragma unroll
  for (int v = 0; v < 8; ++v) { best[v] = 3.4e38f; bidx[v] = 0x7fffffff; }

  // Prime the pipeline: stage tile 0 into buffer 0.
  stage_tile(cb + (size_t)(wave * 16) * DIMC, lds_buf0, lane);

  for (int it = 0; it < 16; ++it) {
    const int colbase = it * 128 + wave * 16;
    const int col = colbase + lrow;
    const unsigned lds_cur = (it & 1) ? lds_buf1 : lds_buf0;
    const unsigned lds_nxt = (it & 1) ? lds_buf0 : lds_buf1;

    if (it + 1 < 16) {
      // Issue next tile's async copy, then wait until only it is in flight
      // (in-order completion => current tile has fully landed in LDS).
      stage_tile(cb + (size_t)(colbase + 128) * DIMC, lds_nxt, lane);
      if (it + 2 < 16)  // warm L2 two tiles ahead
        __builtin_prefetch(cb + (size_t)(colbase + 256 + lrow) * DIMC, 0, 0);
      asm volatile("s_wait_asynccnt 0x18" ::: "memory");   // <= 24 outstanding
    } else {
      asm volatile("s_wait_asynccnt 0x0" ::: "memory");    // drain last tile
    }

    // B reads from LDS (ds_load_b128 pairs): lane holds col lrow, 16 contig K.
    const __bf16* bbuf = (const __bf16*)(&bstage[wave][it & 1][0]) + lrow * DIMC + 16 * kg;
    (void)lds_cur;

    v8f acc0 = {};
    v8f acc1 = {};
#pragma unroll
    for (int c = 0; c < 12; c += 2) {   // two independent accumulator chains
      v16bf b0 = *(const v16bf*)(bbuf + (c + 0) * 32);
      v16bf b1 = *(const v16bf*)(bbuf + (c + 1) * 32);
      acc0 = __builtin_amdgcn_wmma_f32_16x16x32_bf16(
          false, A[c + 0], false, b0, (short)0, acc0, false, false);
      acc1 = __builtin_amdgcn_wmma_f32_16x16x32_bf16(
          false, A[c + 1], false, b1, (short)0, acc1, false, false);
    }

    const float e2c = e2[col];
#pragma unroll
    for (int v = 0; v < 8; ++v) {       // lane holds rows v + 8*kg, column col
      const float s = e2c - 2.0f * (acc0[v] + acc1[v]);
      if (s < best[v]) { best[v] = s; bidx[v] = col; }
    }
  }

  // ---- reduce (score,index) across the 16 lanes that share a row ----------
#pragma unroll
  for (int v = 0; v < 8; ++v) {
    float s = best[v];
    int   i = bidx[v];
#pragma unroll
    for (int off = 8; off > 0; off >>= 1) {
      const float os = __shfl_xor(s, off, 32);
      const int   oi = __shfl_xor(i, off, 32);
      if (os < s || (os == s && oi < i)) { s = os; i = oi; }
    }
    if (lrow == 0) {                    // lanes 0 and 16 own rows v, v+8
      const int row = v + 8 * kg;
      lds_s[wave * 16 + row] = s;
      lds_i[wave * 16 + row] = i;
    }
  }
  __syncthreads();

  // ---- reduce across the 8 waves, write final argmin ----------------------
  if (threadIdx.x < 16) {
    const int row = threadIdx.x;
    float s = lds_s[row];
    int   i = lds_i[row];
    for (int w = 1; w < 8; ++w) {
      const float os = lds_s[w * 16 + row];
      const int   oi = lds_i[w * 16 + row];
      if (os < s || (os == s && oi < i)) { s = os; i = oi; }
    }
    min_idx[rowbase + row] = i;
  }
}

// ---------------------------------------------------------------------------
// 4) Gather z_q (forward of straight-through = codebook[idx]) into NCHW and
//    accumulate per-block partial loss sums (f32, deterministic order).
//    Output linear index == input linear index, so both sides are coalesced.
// ---------------------------------------------------------------------------
__global__ __launch_bounds__(256) void gather_loss(const float* __restrict__ z,
                                                   const float* __restrict__ cb,
                                                   const int* __restrict__ min_idx,
                                                   float* __restrict__ out,
                                                   float* __restrict__ partials) {
  __shared__ float red[256];
  const int i   = blockIdx.x * 256 + threadIdx.x;     // < OUT_N (exact multiple)
  const int b   = i / (DIMC * HW);
  const int rem = i % (DIMC * HW);
  const int c   = rem / HW;
  const int hw  = rem % HW;
  const int n   = b * HW + hw;
  const int idx = min_idx[n];
  const float q = cb[(size_t)idx * DIMC + c];
  out[i] = q;
  const float d = q - z[i];
  red[threadIdx.x] = d * d;
  __syncthreads();
  for (int off = 128; off > 0; off >>= 1) {
    if (threadIdx.x < off) red[threadIdx.x] += red[threadIdx.x + off];
    __syncthreads();
  }
  if (threadIdx.x == 0) partials[blockIdx.x] = red[0];
}

// ---------------------------------------------------------------------------
// 5) Deterministic final reduction: q_loss = (1 + BETA) * mean(diff^2)
// ---------------------------------------------------------------------------
__global__ __launch_bounds__(256) void finalize_loss(const float* __restrict__ partials,
                                                     int nparts,
                                                     float* __restrict__ out_loss) {
  __shared__ float red[256];
  float s = 0.f;
  for (int i = threadIdx.x; i < nparts; i += 256) s += partials[i];
  red[threadIdx.x] = s;
  __syncthreads();
  for (int off = 128; off > 0; off >>= 1) {
    if (threadIdx.x < off) red[threadIdx.x] += red[threadIdx.x + off];
    __syncthreads();
  }
  if (threadIdx.x == 0)
    *out_loss = red[0] * ((1.0f + BETA) / (float)OUT_N);
}

// ---------------------------------------------------------------------------
extern "C" void kernel_launch(void* const* d_in, const int* in_sizes, int n_in,
                              void* d_out, int out_size, void* d_ws, size_t ws_size,
                              hipStream_t stream) {
  const float* z  = (const float*)d_in[0];   // (32,384,28,28)
  const float* cb = (const float*)d_in[1];   // (2048,384)
  float* out = (float*)d_out;                // z_q flat (OUT_N) ++ q_loss (1)

  char* ws = (char*)d_ws;                    // needs ~21.1 MB
  __bf16* zt_bf = (__bf16*)(ws + WS_ZT);
  __bf16* cb_bf = (__bf16*)(ws + WS_CB);
  float*  e2    = (float*)(ws + WS_E2);
  int*    midx  = (int*)(ws + WS_IDX);
  float*  parts = (float*)(ws + WS_PART);

  prep_codebook<<<KCODE, 128, 0, stream>>>(cb, cb_bf, e2);

  dim3 tb(32, 8);
  dim3 tg(25, 12, BATCH);                    // hw-tiles, c-tiles, batch
  prep_z<<<tg, tb, 0, stream>>>(z, zt_bf);

  vq_argmin<<<NTOK / 16, 256, 0, stream>>>(zt_bf, cb_bf, e2, midx);

  gather_loss<<<OUT_N / 256, 256, 0, stream>>>(z, cb, midx, out, parts);

  finalize_loss<<<1, 256, 0, stream>>>(parts, OUT_N / 256, out + OUT_N);
}
